// TransfConv_64862596104464
// MI455X (gfx1250) — matrix operand order repaired and checked by reading, not compiled, not run on hardware
//
#include <hip/hip_runtime.h>
#include <hip/hip_bf16.h>

// -------- constants from the reference --------
#define NN   100000
#define EE   1200000
#define GG   128
#define F_IN 16
#define HH   64
#define OUTD 5
#define NHID 3

typedef __attribute__((ext_vector_type(16))) _Float16 v16h;
typedef __attribute__((ext_vector_type(8)))  _Float16 v8h;
typedef __attribute__((ext_vector_type(4)))  _Float16 v4h;
typedef __attribute__((ext_vector_type(8)))  float    v8f;

// ---- ordered-int encoding for float atomic max (monotone map to u32) ----
__device__ __forceinline__ unsigned enc_f(float f) {
    unsigned u = __float_as_uint(f);
    return (u & 0x80000000u) ? ~u : (u | 0x80000000u);
}
__device__ __forceinline__ float dec_f(unsigned u) {
    unsigned v = (u & 0x80000000u) ? (u & 0x7fffffffu) : ~u;
    return __uint_as_float(v);
}

// =====================================================================
// Fused QKVS GEMM:  out[n,256] = X[n,kin] @ [Wq|Wk|Wv|Ws] + [bq|bk|bv|bs]
// Block = 512 threads (16 waves, wave32), 64 rows x 256 cols per block.
// - weights staged TRANSPOSED in LDS: B fragment = contiguous 32B run
// - float4 global loads for staging, f16 conversion in registers
// - C slabs staged via LDS; uniform fast-path epilogue (only the last
//   block takes the guarded path)
// =====================================================================
__global__ __launch_bounds__(512) void gnn_gemm_qkvs(
    const float* __restrict__ X, int n, int kin,
    const float* __restrict__ Wq, const float* __restrict__ Wk,
    const float* __restrict__ Wv, const float* __restrict__ Ws,
    const float* __restrict__ bq, const float* __restrict__ bk,
    const float* __restrict__ bv, const float* __restrict__ bs,
    float* __restrict__ out)
{
    __shared__ _Float16 Alds[64][64];     //  8 KB, Alds[row][k]
    __shared__ _Float16 Wt[256][64];      // 32 KB, Wt[col][k] (transposed)
    __shared__ float    blds[256];        //  1 KB
    __shared__ float    Clds[16][256];    // 16 KB, one 16-row slab

    const int tid = threadIdx.x;
    const int rbase0 = blockIdx.x * 64;

    // stage weights transposed: float4 coalesced loads, b16 transposed stores
    // slots: 64 K-rows x 64 col-quads = 4096 -> 8 iters
    for (int i = tid; i < 64 * 64; i += 512) {
        int k = i >> 6, c4 = (i & 63) * 4;
        float4 wv = make_float4(0.f, 0.f, 0.f, 0.f);
        if (k < kin) {
            int m = c4 >> 6, cc = c4 & 63;
            const float* Wp = (m == 0) ? Wq : (m == 1) ? Wk : (m == 2) ? Wv : Ws;
            wv = *(const float4*)(Wp + k * 64 + cc);
        }
        Wt[c4 + 0][k] = (_Float16)wv.x;
        Wt[c4 + 1][k] = (_Float16)wv.y;
        Wt[c4 + 2][k] = (_Float16)wv.z;
        Wt[c4 + 3][k] = (_Float16)wv.w;
    }
    if (tid < 256) {
        int m = tid >> 6, cc = tid & 63;
        const float* bp = (m == 0) ? bq : (m == 1) ? bk : (m == 2) ? bv : bs;
        blds[tid] = bp[cc];
    }
    // stage activations: float4 loads, packed 8B DS stores, zero-pad
    // slots: 64 rows x 16 k-quads = 1024 -> 2 iters
    for (int i = tid; i < 64 * 16; i += 512) {
        int r = i >> 4, k0 = (i & 15) * 4;
        int g = rbase0 + r;
        float4 xv = make_float4(0.f, 0.f, 0.f, 0.f);
        if (g < n && k0 < kin) xv = *(const float4*)(X + (size_t)g * kin + k0);
        v4h hv = { (_Float16)xv.x, (_Float16)xv.y, (_Float16)xv.z, (_Float16)xv.w };
        *(v4h*)(&Alds[r][k0]) = hv;
    }
    __syncthreads();

    const int wave = tid >> 5;
    const int lane = tid & 31;
    const int ln = lane & 15;
    const int hi = lane >> 4;           // 0: lanes 0-15, 1: lanes 16-31
    const int col = wave * 16 + ln;     // output column 0..255

    // B fragments: halves 2j,2j+1 hold K = kbase + hi*16 + 2j, +1
    const v16h b0 = *(const v16h*)(&Wt[col][hi * 16]);
    const v16h b1 = *(const v16h*)(&Wt[col][32 + hi * 16]);
    const float bias = blds[col];
    const bool fullblk = (rbase0 + 64 <= n);

    for (int slab = 0; slab < 4; ++slab) {
        const int r = slab * 16 + ln;
        const int ko = hi * 16;
        // A fragment: halves 0..15 = K ko..ko+15 (contiguous)
        v8h a0l = *(const v8h*)(&Alds[r][ko]);
        v8h a0h = *(const v8h*)(&Alds[r][8 + ko]);
        v8h a1l = *(const v8h*)(&Alds[r][32 + ko]);
        v8h a1h = *(const v8h*)(&Alds[r][40 + ko]);
        v16h a0 = __builtin_shufflevector(a0l, a0h, 0, 1, 2, 3, 4, 5, 6, 7,
                                          8, 9, 10, 11, 12, 13, 14, 15);
        v16h a1 = __builtin_shufflevector(a1l, a1h, 0, 1, 2, 3, 4, 5, 6, 7,
                                          8, 9, 10, 11, 12, 13, 14, 15);
        v8f c = {};
        c = __builtin_amdgcn_wmma_f32_16x16x32_f16(false, a0, false, b0,
                                                   (short)0, c, false, false);
        c = __builtin_amdgcn_wmma_f32_16x16x32_f16(false, a1, false, b1,
                                                   (short)0, c, false, false);
        // stage C slab in LDS (C layout: VGPR p -> row p + hi*8, col = N lane)
        #pragma unroll
        for (int p = 0; p < 8; ++p)
            Clds[p + hi * 8][col] = c[p] + bias;
        __syncthreads();
        // coalesced float4 stores; uniform fast path for full blocks
        if (fullblk) {
            #pragma unroll
            for (int it = 0; it < 2; ++it) {
                int f = tid + it * 512;
                int rr = f >> 6, c4 = f & 63;
                int grow = rbase0 + slab * 16 + rr;
                *(float4*)(out + (size_t)grow * 256 + c4 * 4) =
                    *(const float4*)(&Clds[rr][c4 * 4]);
            }
        } else {
            for (int f = tid; f < 16 * 64; f += 512) {
                int rr = f >> 6, c4 = f & 63;
                int grow = rbase0 + slab * 16 + rr;
                if (grow < n)
                    *(float4*)(out + (size_t)grow * 256 + c4 * 4) =
                        *(const float4*)(&Clds[rr][c4 * 4]);
            }
        }
        __syncthreads();
    }
}

// =====================================================================
// Edge pass A: score[e] = dot(q[dst], k[src]) / 8 ; segment-max into menc.
// 8 lanes per edge, float4 loads, shuffle reduction within groups of 8.
// =====================================================================
__global__ __launch_bounds__(256) void gnn_edge_score(
    const long long* __restrict__ ei, const float* __restrict__ qkvs,
    float* __restrict__ score, unsigned* __restrict__ menc)
{
    int t = blockIdx.x * blockDim.x + threadIdx.x;
    int e = t >> 3, r = t & 7;
    if (e >= EE) return;
    long long s = ei[e], d = ei[EE + e];
    const float* qp = qkvs + (size_t)d * 256 + r * 8;        // q at cols 0..63
    const float* kp = qkvs + (size_t)s * 256 + 64 + r * 8;   // k at cols 64..127
    float4 qa = *(const float4*)qp, qb = *(const float4*)(qp + 4);
    float4 ka = *(const float4*)kp, kb = *(const float4*)(kp + 4);
    float p = qa.x * ka.x + qa.y * ka.y + qa.z * ka.z + qa.w * ka.w +
              qb.x * kb.x + qb.y * kb.y + qb.z * kb.z + qb.w * kb.w;
    p += __shfl_xor(p, 1, 8);
    p += __shfl_xor(p, 2, 8);
    p += __shfl_xor(p, 4, 8);
    if (r == 0) {
        float sc = p * 0.125f;      // 1/sqrt(64)
        score[e] = sc;
        atomicMax(menc + d, enc_f(sc));
    }
}

// =====================================================================
// Edge pass B: ee = exp(score - m[dst]); denom[dst]+=ee; agg[dst]+=ee*v[src]
// =====================================================================
__global__ __launch_bounds__(256) void gnn_edge_agg(
    const long long* __restrict__ ei, const float* __restrict__ qkvs,
    const float* __restrict__ score, const unsigned* __restrict__ menc,
    float* __restrict__ denom, float* __restrict__ aggU)
{
    int t = blockIdx.x * blockDim.x + threadIdx.x;
    int e = t >> 3, r = t & 7;
    if (e >= EE) return;
    long long s = ei[e], d = ei[EE + e];
    float m = dec_f(menc[d]);
    float ee = __expf(score[e] - m);
    if (r == 0) atomicAdd(denom + d, ee);
    const float* vp = qkvs + (size_t)s * 256 + 128 + r * 8;  // v at cols 128..191
    float4 va = *(const float4*)vp, vb = *(const float4*)(vp + 4);
    float* ap = aggU + (size_t)d * 64 + r * 8;
    atomicAdd(ap + 0, ee * va.x);
    atomicAdd(ap + 1, ee * va.y);
    atomicAdd(ap + 2, ee * va.z);
    atomicAdd(ap + 3, ee * va.w);
    atomicAdd(ap + 4, ee * vb.x);
    atomicAdd(ap + 5, ee * vb.y);
    atomicAdd(ap + 6, ee * vb.z);
    atomicAdd(ap + 7, ee * vb.w);
}

// =====================================================================
// Node finish: h[i] = agg[i] / (denom[i]+1e-16) + skip[i]   (skip at 192..255)
// =====================================================================
__global__ __launch_bounds__(256) void gnn_node_finish(
    const float* __restrict__ qkvs, const float* __restrict__ aggU,
    const float* __restrict__ denom, float* __restrict__ hout)
{
    int t = blockIdx.x * blockDim.x + threadIdx.x;
    int i = t >> 4, f4 = t & 15;
    if (i >= NN) return;
    float inv = 1.0f / (denom[i] + 1e-16f);
    float4 ag = *(const float4*)(aggU + (size_t)i * 64 + f4 * 4);
    float4 sk = *(const float4*)(qkvs + (size_t)i * 256 + 192 + f4 * 4);
    float4 o;
    o.x = ag.x * inv + sk.x;
    o.y = ag.y * inv + sk.y;
    o.z = ag.z * inv + sk.z;
    o.w = ag.w * inv + sk.w;
    *(float4*)(hout + (size_t)i * 64 + f4 * 4) = o;
}

// =====================================================================
// Mean pool accumulation
// =====================================================================
__global__ __launch_bounds__(256) void gnn_pool(
    const float* __restrict__ h, const long long* __restrict__ batch,
    float* __restrict__ sums, float* __restrict__ counts)
{
    int t = blockIdx.x * blockDim.x + threadIdx.x;
    int i = t >> 4, f4 = t & 15;
    if (i >= NN) return;
    int g = (int)batch[i];
    float4 v = *(const float4*)(h + (size_t)i * 64 + f4 * 4);
    float* sp = sums + g * 64 + f4 * 4;
    atomicAdd(sp + 0, v.x);
    atomicAdd(sp + 1, v.y);
    atomicAdd(sp + 2, v.z);
    atomicAdd(sp + 3, v.w);
    if (f4 == 0) atomicAdd(counts + g, 1.0f);
}

// =====================================================================
// Head: out[G,5] = (sums / max(counts,1)) @ Wf + bf
// =====================================================================
__global__ __launch_bounds__(640) void gnn_head(
    const float* __restrict__ sums, const float* __restrict__ counts,
    const float* __restrict__ Wf, const float* __restrict__ bf,
    float* __restrict__ out)
{
    int t = threadIdx.x;
    int g = t / OUTD, o = t % OUTD;
    if (g >= GG) return;
    float inv = 1.0f / fmaxf(counts[g], 1.0f);
    float acc = bf[o];
    for (int f = 0; f < HH; ++f)
        acc += (sums[g * 64 + f] * inv) * Wf[f * OUTD + o];
    out[g * OUTD + o] = acc;
}

// =====================================================================
extern "C" void kernel_launch(void* const* d_in, const int* in_sizes, int n_in,
                              void* d_out, int out_size, void* d_ws, size_t ws_size,
                              hipStream_t stream) {
    (void)in_sizes; (void)n_in; (void)out_size; (void)ws_size;

    const float*     x   = (const float*)d_in[0];
    const long long* ei  = (const long long*)d_in[1];   // [2,E] int64
    const long long* bid = (const long long*)d_in[2];   // [N]   int64
    const float *Wq0 = (const float*)d_in[3],  *bq0 = (const float*)d_in[4];
    const float *Wk0 = (const float*)d_in[5],  *bk0 = (const float*)d_in[6];
    const float *Wv0 = (const float*)d_in[7],  *bv0 = (const float*)d_in[8];
    const float *Ws0 = (const float*)d_in[9],  *bs0 = (const float*)d_in[10];
    const float *Wqh = (const float*)d_in[11], *bqh = (const float*)d_in[12];
    const float *Wkh = (const float*)d_in[13], *bkh = (const float*)d_in[14];
    const float *Wvh = (const float*)d_in[15], *bvh = (const float*)d_in[16];
    const float *Wsh = (const float*)d_in[17], *bsh = (const float*)d_in[18];
    const float *Wf  = (const float*)d_in[19], *bf  = (const float*)d_in[20];
    float* out = (float*)d_out;

    // ---- workspace carve-up (floats) ----
    float*    ws    = (float*)d_ws;
    float*    h     = ws;                         // N*64
    float*    qkvs  = h     + (size_t)NN * 64;    // N*256
    float*    score = qkvs  + (size_t)NN * 256;   // E
    float*    zbase = score + (size_t)EE;         // zero-region start
    unsigned* menc  = (unsigned*)zbase;           // N
    float*    denom = zbase + NN;                 // N
    float*    aggU  = denom + NN;                 // N*64
    float*    sums  = aggU  + (size_t)NN * 64;    // G*64
    float*    cnts  = sums  + (size_t)GG * 64;    // G

    const size_t zbytes = (size_t)(NN + NN + NN * 64) * sizeof(float);

    const int gemm_grid = (NN + 63) / 64;
    const int edge_grid = (EE * 8 + 255) / 256;
    const int node_grid = (NN * 16 + 255) / 256;

    const float* hin = x;
    int kin = F_IN;
    const float *Wq = Wq0, *Wk = Wk0, *Wv = Wv0, *Ws = Ws0;
    const float *bq = bq0, *bk = bk0, *bv = bv0, *bs = bs0;

    for (int layer = 0; layer < 1 + NHID; ++layer) {
        gnn_gemm_qkvs<<<gemm_grid, 512, 0, stream>>>(
            hin, NN, kin, Wq, Wk, Wv, Ws, bq, bk, bv, bs, qkvs);
        hipMemsetAsync(zbase, 0, zbytes, stream);
        gnn_edge_score<<<edge_grid, 256, 0, stream>>>(ei, qkvs, score, menc);
        gnn_edge_agg<<<edge_grid, 256, 0, stream>>>(ei, qkvs, score, menc, denom, aggU);
        gnn_node_finish<<<node_grid, 256, 0, stream>>>(qkvs, aggU, denom, h);

        // next layer config
        hin = h; kin = HH;
        if (layer < NHID) {
            Wq = Wqh + (size_t)layer * HH * HH;  bq = bqh + (size_t)layer * HH;
            Wk = Wkh + (size_t)layer * HH * HH;  bk = bkh + (size_t)layer * HH;
            Wv = Wvh + (size_t)layer * HH * HH;  bv = bvh + (size_t)layer * HH;
            Ws = Wsh + (size_t)layer * HH * HH;  bs = bsh + (size_t)layer * HH;
        }
    }

    hipMemsetAsync(sums, 0, (size_t)(GG * 64 + GG) * sizeof(float), stream);
    gnn_pool<<<node_grid, 256, 0, stream>>>(h, bid, sums, cnts);
    gnn_head<<<1, 640, 0, stream>>>(sums, cnts, Wf, bf, out);
}
